// MultiHeadCrossAttention_70952859730436
// MI455X (gfx1250) — matrix operand order repaired and checked by reading, not compiled, not run on hardware
//
#include <hip/hip_runtime.h>
#include <hip/hip_bf16.h>
#include <stdint.h>

// ---------------- problem constants ----------------
static constexpr int kB     = 16;
static constexpr int kC     = 256;
static constexpr int kHW    = 4096;   // 64*64
static constexpr int kRD    = 256;
static constexpr int kHEADS = 8;

typedef __bf16 bf16_t;
typedef __attribute__((ext_vector_type(16))) __bf16 v16bf;
typedef __attribute__((ext_vector_type(8)))  float  v8f;
typedef __attribute__((ext_vector_type(4)))  unsigned tv4u;
typedef __attribute__((ext_vector_type(8)))  int      tv8i;
typedef __attribute__((ext_vector_type(4)))  int      tv4i;

// CDNA5 data movers (guarded so a missing builtin degrades, never breaks)
#if __has_builtin(__builtin_amdgcn_global_load_async_to_lds_b128) && \
    __has_builtin(__builtin_amdgcn_s_wait_asynccnt)
#define USE_ASYNC_LDS 1
#endif
#if __has_builtin(__builtin_amdgcn_tensor_load_to_lds) && \
    __has_builtin(__builtin_amdgcn_s_wait_tensorcnt)
#define USE_TDM 1
#endif

// async-to-LDS builtin takes (v4i AS1*, v4i AS3*, imm offset, imm cpol)
#define AS1P(p) ((__attribute__((address_space(1))) tv4i*)(uintptr_t)(p))
#define AS3P(p) ((__attribute__((address_space(3))) tv4i*)(uintptr_t)(p))

// LDS tile pitches (padded to de-conflict 64-bank LDS on fragment reads)
static constexpr int kPF = 68;   // f32 tile pitch  (gram): 68*4B  -> bank m*4+x
static constexpr int kPB = 72;   // bf16 tile pitch (pmat): 72*2B = 144B (16B-mult)

// ---------------- WMMA helpers ----------------
__device__ __forceinline__ v8f wmma_bf16(v16bf a, v16bf b, v8f c) {
  return __builtin_amdgcn_wmma_f32_16x16x32_bf16(
      false, a, false, b, (short)0, c, false, false);
}

// A fragment, 16x32 bf16, row-major source.
// lanes 0-15 (half=0): V0..3=K0..7,  V4..7=K16..23
// lanes16-31 (half=1): V0..3=K8..15, V4..7=K24..31
__device__ __forceinline__ v16bf frag_a(const bf16_t* base, int stride, int ks) {
  int lane = threadIdx.x & 31;
  int half = lane >> 4, m = lane & 15;
  const bf16_t* r = base + m * stride;
  v16bf f;
#pragma unroll
  for (int i = 0; i < 8; ++i) f[i] = r[ks + half * 8 + i];
#pragma unroll
  for (int i = 0; i < 8; ++i) f[8 + i] = r[ks + 16 + half * 8 + i];
  return f;
}
// B fragment, 32x16 bf16, B[k][n] = src_row(n)[k].
// lanes 0-15: V0..3=K0..7, V4..7=K16..23; lanes 16-31: K8..15, K24..31.
__device__ __forceinline__ v16bf frag_b(const bf16_t* base, int stride, int ks) {
  int lane = threadIdx.x & 31;
  int khalf = lane >> 4, n = lane & 15;
  const bf16_t* r = base + n * stride;
  v16bf f;
#pragma unroll
  for (int i = 0; i < 8; ++i) f[i] = r[ks + khalf * 8 + i];
#pragma unroll
  for (int i = 0; i < 8; ++i) f[8 + i] = r[ks + 16 + khalf * 8 + i];
  return f;
}
// Same fragments but from f32 LDS tiles, converting on the fly.
__device__ __forceinline__ v16bf frag_a_f32(const float* base, int stride, int ks) {
  int lane = threadIdx.x & 31;
  int half = lane >> 4, m = lane & 15;
  const float* r = base + m * stride;
  v16bf f;
#pragma unroll
  for (int i = 0; i < 8; ++i) f[i] = (bf16_t)r[ks + half * 8 + i];
#pragma unroll
  for (int i = 0; i < 8; ++i) f[8 + i] = (bf16_t)r[ks + 16 + half * 8 + i];
  return f;
}
__device__ __forceinline__ v16bf frag_b_f32(const float* base, int stride, int ks) {
  int lane = threadIdx.x & 31;
  int khalf = lane >> 4, n = lane & 15;
  const float* r = base + n * stride;
  v16bf f;
#pragma unroll
  for (int i = 0; i < 8; ++i) f[i] = (bf16_t)r[ks + khalf * 8 + i];
#pragma unroll
  for (int i = 0; i < 8; ++i) f[8 + i] = (bf16_t)r[ks + 16 + khalf * 8 + i];
  return f;
}

// ---------------- K0: prep ----------------
__global__ __launch_bounds__(256) void prep_kernel(
    const float* __restrict__ Wq, const float* __restrict__ Wk,
    const float* __restrict__ Wv, const float* __restrict__ bv,
    bf16_t* __restrict__ Wqb, bf16_t* __restrict__ Wkb,
    float* __restrict__ WvSum, float* __restrict__ bvSum) {
  int i = blockIdx.x * blockDim.x + threadIdx.x;
  if (i < kRD * kC) {
    Wqb[i] = (bf16_t)Wq[i];
    Wkb[i] = (bf16_t)Wk[i];
  }
  if (i < kHEADS * kC) {
    int h = i / kC, c = i % kC;
    float s = 0.f;
    for (int q = 0; q < 32; ++q) s += Wv[(h * 32 + q) * kC + c];
    WvSum[h * kC + c] = s;
  }
  if (i < kHEADS) {
    float s = 0.f;
    for (int q = 0; q < 32; ++q) s += bv[i * 32 + q];
    bvSum[i] = s;
  }
}

// ---------------- K1: Gram  G[b] = x1[b] * x2[b]^T  (C x C, K = HW) ----------
// Tile staging via Tensor Data Mover (2D D#, LDS padding 64dw+4dw -> pitch 68).
__global__ __launch_bounds__(256) void gram_kernel(
    const float* __restrict__ x1, const float* __restrict__ x2,
    bf16_t* __restrict__ G, bf16_t* __restrict__ GT,
    float* __restrict__ t1, float* __restrict__ t2) {
  __shared__ __align__(16) float lAf[64 * kPF];
  __shared__ __align__(16) float lBf[64 * kPF];
  int blk = blockIdx.x;
  int b = blk / 16;
  int rt = (blk % 16) / 4, ct = blk % 4;
  int r0 = rt * 64, c0 = ct * 64;
  int tid = threadIdx.x;
  int w = tid >> 5;
  int rsub = (w & 3) * 16, csub = (w >> 2) * 32;
  v8f acc0 = {}; v8f acc1 = {};
  float rs = 0.f;
  const float* A0 = x1 + (size_t)b * kC * kHW;
  const float* B0 = x2 + (size_t)b * kC * kHW;

#if defined(USE_TDM)
  // D# group1: data_size=4B(2), pad_enable, pad_interval=64dw(5), pad_amount=4dw(3),
  // tensor_dim0=4096, tensor_dim1=256, tile=64x64, dim0_stride=4096
  tv8i g1 = {};
  g1[0] = (int)((2u << 16) | (1u << 20) | (5u << 22) | (3u << 25));
  g1[1] = (int)((4096u & 0xFFFFu) << 16);                 // tensor_dim0 lo16
  g1[2] = (int)((4096u >> 16) | ((256u & 0xFFFFu) << 16)); // td0 hi16 | td1 lo16
  g1[3] = (int)((256u >> 16) | (64u << 16));               // td1 hi16 | tile_dim0
  g1[4] = 64;                                              // tile_dim1 (tile_dim2=0)
  g1[5] = 4096;                                            // dim0_stride lo32
  g1[6] = 0; g1[7] = 0;
  tv4i z4 = {};
  unsigned ldsA = (unsigned)(uintptr_t)&lAf[0];
  unsigned ldsB = (unsigned)(uintptr_t)&lBf[0];
#endif

  for (int k0 = 0; k0 < kHW; k0 += 64) {
#if defined(USE_TDM)
    if (w < 2) {
      const float* src = (w == 0) ? (A0 + (size_t)r0 * kHW + k0)
                                  : (B0 + (size_t)c0 * kHW + k0);
      uint64_t ga = (uint64_t)(uintptr_t)src;
      tv4u g0;
      g0[0] = 1u;                                          // count=1 (valid D#)
      g0[1] = (w == 0) ? ldsA : ldsB;                      // lds_addr
      g0[2] = (unsigned)ga;                                // global_addr lo
      g0[3] = (unsigned)((ga >> 32) & 0x01FFFFFFu) | (2u << 30);  // ga hi | type=2
#if __clang_major__ >= 23
      tv8i gx = {};
      __builtin_amdgcn_tensor_load_to_lds(g0, g1, z4, z4, gx, 0);
#else
      __builtin_amdgcn_tensor_load_to_lds(g0, g1, z4, z4, 0);
#endif
    }
    __builtin_amdgcn_s_wait_tensorcnt(0);
#elif defined(USE_ASYNC_LDS)
#pragma unroll
    for (int s = 0; s < 4; ++s) {
      int slot = tid + s * 256;          // 1024 16B-chunks per matrix
      int row = slot >> 4, ch = slot & 15;
      __builtin_amdgcn_global_load_async_to_lds_b128(
          AS1P(A0 + (size_t)(r0 + row) * kHW + k0 + ch * 4),
          AS3P(&lAf[row * kPF + ch * 4]), 0, 0);
      __builtin_amdgcn_global_load_async_to_lds_b128(
          AS1P(B0 + (size_t)(c0 + row) * kHW + k0 + ch * 4),
          AS3P(&lBf[row * kPF + ch * 4]), 0, 0);
    }
    __builtin_amdgcn_s_wait_asynccnt(0);
#else
#pragma unroll
    for (int s = 0; s < 4; ++s) {
      int slot = tid + s * 256;
      int row = slot >> 4, ch = slot & 15;
      *(float4*)&lAf[row * kPF + ch * 4] =
          *(const float4*)(A0 + (size_t)(r0 + row) * kHW + k0 + ch * 4);
      *(float4*)&lBf[row * kPF + ch * 4] =
          *(const float4*)(B0 + (size_t)(c0 + row) * kHW + k0 + ch * 4);
    }
#endif
    __syncthreads();
    // fold exact f32 bias row-sums in for free
    if (ct == 0 && tid < 64) {
      float s = 0.f;
      for (int j = 0; j < 64; ++j) s += lAf[tid * kPF + j];
      rs += s;
    }
    if (rt == 0 && tid >= 64 && tid < 128) {
      float s = 0.f; int r = tid - 64;
      for (int j = 0; j < 64; ++j) s += lBf[r * kPF + j];
      rs += s;
    }
#pragma unroll
    for (int ks = 0; ks < 64; ks += 32) {
      v16bf a  = frag_a_f32(&lAf[rsub * kPF], kPF, ks);
      v16bf b0 = frag_b_f32(&lBf[csub * kPF], kPF, ks);
      v16bf b1 = frag_b_f32(&lBf[(csub + 16) * kPF], kPF, ks);
      acc0 = wmma_bf16(a, b0, acc0);
      acc1 = wmma_bf16(a, b1, acc1);
    }
    __syncthreads();
  }
  if (ct == 0 && tid < 64) t1[b * kC + r0 + tid] = rs;
  if (rt == 0 && tid >= 64 && tid < 128) t2[b * kC + c0 + (tid - 64)] = rs;

  int lane = tid & 31, n = lane & 15, mh = (lane >> 4) * 8;
  bf16_t* Gb  = G  + (size_t)b * kC * kC;
  bf16_t* GTb = GT + (size_t)b * kC * kC;
#pragma unroll
  for (int i = 0; i < 8; ++i) {
    int gr  = r0 + rsub + mh + i;
    int gc0 = c0 + csub + n;
    int gc1 = gc0 + 16;
    Gb[gr * kC + gc0]  = (bf16_t)acc0[i];
    Gb[gr * kC + gc1]  = (bf16_t)acc1[i];
    GTb[gc0 * kC + gr] = (bf16_t)acc0[i];
    GTb[gc1 * kC + gr] = (bf16_t)acc1[i];
  }
}

// ---------------- K2: u1[b][q] = Wq_q . t1[b], u2[b][k] = Wk_k . t2[b] -------
__global__ __launch_bounds__(256) void u_kernel(
    const float* __restrict__ Wq, const float* __restrict__ Wk,
    const float* __restrict__ t1, const float* __restrict__ t2,
    float* __restrict__ u1, float* __restrict__ u2) {
  int b = blockIdx.x, q = threadIdx.x;
  const float* t1b = t1 + b * kC;
  const float* t2b = t2 + b * kC;
  float s1 = 0.f, s2 = 0.f;
  for (int c = 0; c < kC; ++c) {
    s1 += Wq[q * kC + c] * t1b[c];
    s2 += Wk[q * kC + c] * t2b[c];
  }
  u1[b * kRD + q] = s1;
  u2[b * kRD + q] = s2;
}

// ---------------- K3a: P[b] = Wq_bf * G[b]   (RD x C, K = C) -----------------
// bf16 tile staging via async global->LDS copies (ASYNCcnt).
__global__ __launch_bounds__(256) void pmat_kernel(
    const bf16_t* __restrict__ Wqb, const bf16_t* __restrict__ GT,
    bf16_t* __restrict__ P) {
  __shared__ __align__(16) bf16_t lA[64 * kPB];
  __shared__ __align__(16) bf16_t lB[64 * kPB];
  int blk = blockIdx.x;
  int b = blk / 16;
  int rt = (blk % 16) / 4, ct = blk % 4;
  int r0 = rt * 64, c0 = ct * 64;
  int tid = threadIdx.x;
  int w = tid >> 5;
  int rsub = (w & 3) * 16, csub = (w >> 2) * 32;
  v8f acc0 = {}; v8f acc1 = {};
  const bf16_t* A0 = Wqb;                       // shared across batches
  const bf16_t* B0 = GT + (size_t)b * kC * kC;  // rows = columns of G

  for (int k0 = 0; k0 < kC; k0 += 64) {
#pragma unroll
    for (int s = 0; s < 2; ++s) {
      int slot = tid + s * 256;        // 512 16B-chunks per matrix
      int row = slot >> 3, u = slot & 7;
#if defined(USE_ASYNC_LDS)
      __builtin_amdgcn_global_load_async_to_lds_b128(
          AS1P(A0 + (size_t)(r0 + row) * kC + k0 + u * 8),
          AS3P(&lA[row * kPB + u * 8]), 0, 0);
      __builtin_amdgcn_global_load_async_to_lds_b128(
          AS1P(B0 + (size_t)(c0 + row) * kC + k0 + u * 8),
          AS3P(&lB[row * kPB + u * 8]), 0, 0);
#else
      *(uint4*)&lA[row * kPB + u * 8] =
          *(const uint4*)(A0 + (size_t)(r0 + row) * kC + k0 + u * 8);
      *(uint4*)&lB[row * kPB + u * 8] =
          *(const uint4*)(B0 + (size_t)(c0 + row) * kC + k0 + u * 8);
#endif
    }
#if defined(USE_ASYNC_LDS)
    __builtin_amdgcn_s_wait_asynccnt(0);
#endif
    __syncthreads();
#pragma unroll
    for (int ks = 0; ks < 64; ks += 32) {
      v16bf a  = frag_a(&lA[rsub * kPB], kPB, ks);
      v16bf b0 = frag_b(&lB[csub * kPB], kPB, ks);
      v16bf b1 = frag_b(&lB[(csub + 16) * kPB], kPB, ks);
      acc0 = wmma_bf16(a, b0, acc0);
      acc1 = wmma_bf16(a, b1, acc1);
    }
    __syncthreads();
  }
  int lane = tid & 31, n = lane & 15, mh = (lane >> 4) * 8;
  bf16_t* Pb = P + (size_t)b * kRD * kC;
#pragma unroll
  for (int i = 0; i < 8; ++i) {
    int pr = r0 + rsub + mh + i;
    Pb[pr * kC + c0 + csub + n]      = (bf16_t)acc0[i];
    Pb[pr * kC + c0 + csub + 16 + n] = (bf16_t)acc1[i];
  }
}

// ---------------- K3b: per-head logits  L[b,h] = P_h * Wk_h^T + bias ---------
__global__ __launch_bounds__(32) void logits_kernel(
    const bf16_t* __restrict__ P, const bf16_t* __restrict__ Wkb,
    const float* __restrict__ bq, const float* __restrict__ bk,
    const float* __restrict__ u1, const float* __restrict__ u2,
    float* __restrict__ Lg) {
  int blk = blockIdx.x;
  int b = blk / kHEADS, h = blk % kHEADS;
  const bf16_t* A0 = P + ((size_t)b * kRD + h * 32) * kC;
  const bf16_t* B0 = Wkb + (size_t)(h * 32) * kC;
  v8f acc[2][2] = {};
  for (int ks = 0; ks < kC; ks += 32) {
    v16bf a0 = frag_a(A0,           kC, ks);
    v16bf a1 = frag_a(A0 + 16 * kC, kC, ks);
    v16bf b0 = frag_b(B0,           kC, ks);
    v16bf b1 = frag_b(B0 + 16 * kC, kC, ks);
    acc[0][0] = wmma_bf16(a0, b0, acc[0][0]);
    acc[0][1] = wmma_bf16(a0, b1, acc[0][1]);
    acc[1][0] = wmma_bf16(a1, b0, acc[1][0]);
    acc[1][1] = wmma_bf16(a1, b1, acc[1][1]);
  }
  int lane = threadIdx.x & 31, n = lane & 15, mh = (lane >> 4) * 8;
  const float inv = 0.17677669529663687f;  // 1/sqrt(32)
  float* Lb = Lg + ((size_t)b * kHEADS + h) * 32 * 32;
#pragma unroll
  for (int rt = 0; rt < 2; ++rt)
#pragma unroll
    for (int ctc = 0; ctc < 2; ++ctc)
#pragma unroll
      for (int i = 0; i < 8; ++i) {
        int q = rt * 16 + mh + i, k = ctc * 16 + n;
        int qg = h * 32 + q, kg = h * 32 + k;
        float v = acc[rt][ctc][i];
        v += bq[qg] * u2[b * kRD + kg] + bk[kg] * u1[b * kRD + qg]
             + (float)kHW * bq[qg] * bk[kg];
        Lb[q * 32 + k] = v * inv;
      }
}

// ---------------- K4: softmax row-sums S, then W2[b][c][h] -------------------
__global__ __launch_bounds__(256) void softmax_w2_kernel(
    const float* __restrict__ Lg, const float* __restrict__ Wo,
    float* __restrict__ W2) {
  __shared__ float S[kHEADS][32];
  int b = blockIdx.x, tid = threadIdx.x;
  int h = tid >> 5, q = tid & 31;
  {
    const float* row = Lg + (((size_t)b * kHEADS + h) * 32 + q) * 32;
    float l[32];
    float m = -1e30f;
    for (int k = 0; k < 32; ++k) { l[k] = row[k]; m = fmaxf(m, l[k]); }
    float s = 0.f;
    for (int k = 0; k < 32; ++k) { l[k] = expf(l[k] - m); s += l[k]; }
    float invs = 1.0f / s, rsum = 0.f;
    for (int k = 0; k < 32; ++k) rsum += l[k] * invs;   // faithful attn row-sum
    S[h][q] = rsum;
  }
  __syncthreads();
  int c = tid;
  const float* wo = Wo + (size_t)c * kRD;
  for (int hh = 0; hh < kHEADS; ++hh) {
    float s = 0.f;
    for (int qq = 0; qq < 32; ++qq) s += wo[hh * 32 + qq] * S[hh][qq];
    W2[((size_t)b * kC + c) * kHEADS + hh] = s;
  }
}

// ---------------- K5: fused  Vsum -> out = W2*Vsum + bo + x4 -----------------
__global__ __launch_bounds__(256) void out_kernel(
    const float* __restrict__ x3, const float* __restrict__ x4,
    const float* __restrict__ WvSum, const float* __restrict__ bvSum,
    const float* __restrict__ W2, const float* __restrict__ bo,
    float* __restrict__ out) {
  __shared__ float sWv[kHEADS * kC];
  __shared__ float sW2[kC * kHEADS];
  __shared__ float sbo[kC];
  __shared__ float sbv[kHEADS];
  int b = blockIdx.x >> 4;
  int chunk = blockIdx.x & 15;
  int tid = threadIdx.x;
  for (int i = tid; i < kHEADS * kC; i += 256) sWv[i] = WvSum[i];
  for (int i = tid; i < kC * kHEADS; i += 256) sW2[i] = W2[(size_t)b * kC * kHEADS + i];
  if (tid < kC) sbo[tid] = bo[tid];
  if (tid < kHEADS) sbv[tid] = bvSum[tid];
  __syncthreads();
  int col = chunk * 256 + tid;
  const float* x3b = x3 + (size_t)b * kC * kHW;
  const float* x4b = x4 + (size_t)b * kC * kHW;
  float* ob = out + (size_t)b * kC * kHW;
  float vs[kHEADS];
#pragma unroll
  for (int h = 0; h < kHEADS; ++h) vs[h] = sbv[h];
  for (int c = 0; c < kC; ++c) {
    const float* p = x3b + (size_t)c * kHW + col;
    if (c + 1 < kC) __builtin_prefetch(p + kHW, 0, 1);
    float xv = *p;
#pragma unroll
    for (int h = 0; h < kHEADS; ++h) vs[h] += sWv[h * kC + c] * xv;
  }
  for (int c = 0; c < kC; ++c) {
    float o = sbo[c] + x4b[(size_t)c * kHW + col];
    const float* w2c = &sW2[c * kHEADS];
#pragma unroll
    for (int h = 0; h < kHEADS; ++h) o += w2c[h] * vs[h];
    ob[(size_t)c * kHW + col] = o;
  }
}

// ---------------- launch ----------------
extern "C" void kernel_launch(void* const* d_in, const int* in_sizes, int n_in,
                              void* d_out, int out_size, void* d_ws, size_t ws_size,
                              hipStream_t stream) {
  const float* x1 = (const float*)d_in[0];
  const float* x2 = (const float*)d_in[1];
  const float* x3 = (const float*)d_in[2];
  const float* x4 = (const float*)d_in[3];
  const float* Wq = (const float*)d_in[4];
  const float* bq = (const float*)d_in[5];
  const float* Wk = (const float*)d_in[6];
  const float* bk = (const float*)d_in[7];
  const float* Wv = (const float*)d_in[8];
  const float* bv = (const float*)d_in[9];
  const float* Wo = (const float*)d_in[10];
  const float* bo = (const float*)d_in[11];
  float* out = (float*)d_out;

  char* ws = (char*)d_ws;
  size_t off = 0;
  auto alloc = [&](size_t bytes) -> char* {
    char* p = ws + off;
    off = (off + bytes + 255) & ~((size_t)255);
    return p;
  };
  bf16_t* Wqb   = (bf16_t*)alloc((size_t)kRD * kC * sizeof(bf16_t));
  bf16_t* Wkb   = (bf16_t*)alloc((size_t)kRD * kC * sizeof(bf16_t));
  float*  WvSum = (float*)alloc((size_t)kHEADS * kC * sizeof(float));
  float*  bvSum = (float*)alloc((size_t)kHEADS * sizeof(float));
  float*  t1    = (float*)alloc((size_t)kB * kC * sizeof(float));
  float*  t2    = (float*)alloc((size_t)kB * kC * sizeof(float));
  float*  u1    = (float*)alloc((size_t)kB * kRD * sizeof(float));
  float*  u2    = (float*)alloc((size_t)kB * kRD * sizeof(float));
  bf16_t* G     = (bf16_t*)alloc((size_t)kB * kC * kC * sizeof(bf16_t));
  bf16_t* GT    = (bf16_t*)alloc((size_t)kB * kC * kC * sizeof(bf16_t));
  bf16_t* P     = (bf16_t*)alloc((size_t)kB * kRD * kC * sizeof(bf16_t));
  float*  Lg    = (float*)alloc((size_t)kB * kHEADS * 32 * 32 * sizeof(float));
  float*  W2    = (float*)alloc((size_t)kB * kC * kHEADS * sizeof(float));

  prep_kernel<<<256, 256, 0, stream>>>(Wq, Wk, Wv, bv, Wqb, Wkb, WvSum, bvSum);
  gram_kernel<<<kB * 16, 256, 0, stream>>>(x1, x2, G, GT, t1, t2);
  u_kernel<<<kB, 256, 0, stream>>>(Wq, Wk, t1, t2, u1, u2);
  pmat_kernel<<<kB * 16, 256, 0, stream>>>(Wqb, GT, P);
  logits_kernel<<<kB * kHEADS, 32, 0, stream>>>(P, Wkb, bq, bk, u1, u2, Lg);
  softmax_w2_kernel<<<kB, 256, 0, stream>>>(Lg, Wo, W2);
  out_kernel<<<kB * 16, 256, 0, stream>>>(x3, x4, WvSum, bvSum, W2, bo, out);
}
